// MambaGraphBlock_77678778515584
// MI455X (gfx1250) — matrix operand order repaired and checked by reading, not compiled, not run on hardware
//
#include <hip/hip_runtime.h>

typedef __bf16 bf16;
typedef __attribute__((ext_vector_type(16))) __bf16 bf16x16;
typedef __attribute__((ext_vector_type(8)))  float  floatx8;
typedef __attribute__((ext_vector_type(4)))  int    intx4;

struct __align__(16) U4 { unsigned u[4]; };
union Frag { bf16x16 v; U4 q[2]; };

#define LDSS 40   // LDS row stride in bf16: 80B = 5*16B (16B-aligned vectors, staggers banks)

#ifndef __has_builtin
#define __has_builtin(x) 0
#endif
#if __has_builtin(__builtin_amdgcn_global_load_async_to_lds_b128) && \
    __has_builtin(__builtin_amdgcn_s_wait_asynccnt)
#define HAVE_ASYNC_LDS 1
// param0 is int4 in global AS (prints as "int __vector(4) __device__*"),
// param1 is the LDS-side pointer (addrspace(3)).
#define TO_GLB(p) ((__attribute__((address_space(1))) intx4*)(p))
#define TO_LDS(p) ((__attribute__((address_space(3))) intx4*)(p))
#else
#define HAVE_ASYNC_LDS 0
#endif

// ---------------------------------------------------------------------------
// Generic bf16 WMMA GEMM: C[M,N] = epilogue(A[M,K] @ B)  (fp32 accumulate)
//   B_TRANS=1: B is weight [N,K] row-major (A @ B^T);  B_TRANS=0: B is [K,N]
//   ACT: 0=none, 1=exact GELU
//   REMAP: row m=(bn*T+t) -> output row (b*T+t)*N+node (mamba out_proj)
//   HAS_RESID: out = resid + gate * (acc [+bias][act])
//   ALIGNED: M%128==0 && N%128==0 -> unguarded loads + double-buffered
//            pipeline (async global->LDS when toolchain exposes it)
// Block: 256 threads (8 waves), tile 128x128, K-step 32.
// ---------------------------------------------------------------------------
template<bool B_TRANS, int ACT, bool STORE_BF16, bool REMAP, bool HAS_RESID, bool ALIGNED>
__global__ __launch_bounds__(256) void gemm_bf16_wmma(
    const bf16* __restrict__ A, const bf16* __restrict__ Bm,
    const float* __restrict__ bias, const float* __restrict__ resid,
    const float* __restrict__ gate_ptr, void* __restrict__ Cout,
    int M, int N, int K,
    long long batchA, long long batchB, long long batchC,
    int Tdim, int Nnodes)
{
  __shared__ __align__(16) bf16 As[2][128 * LDSS];
  __shared__ __align__(16) bf16 Bs[2][128 * LDSS];
  const int tid   = threadIdx.x;
  const int lane  = tid & 31;
  const int wid   = tid >> 5;
  const int khalf = lane >> 4;   // wave32: lanes 0-15 / 16-31
  const int l15   = lane & 15;
  const long long bz = blockIdx.z;
  const bf16* Ab = A + bz * batchA;
  const bf16* Bb = Bm + bz * batchB;
  const int m0  = blockIdx.y * 128;
  const int n0  = blockIdx.x * 128;
  const int wm0 = (wid & 1) * 64;
  const int wn0 = (wid >> 1) * 32;

  floatx8 acc[4][2];
#pragma unroll
  for (int i = 0; i < 4; ++i)
#pragma unroll
    for (int j = 0; j < 2; ++j)
#pragma unroll
      for (int r = 0; r < 8; ++r) acc[i][j][r] = 0.0f;

  // ---- staging helpers (256 threads move 128x32 bf16 = 8KB per tile) ----
  auto loadA_regs = [&](int kt, U4* r) {
#pragma unroll
    for (int i = 0; i < 2; ++i) {
      int idx = tid + i * 256, row = idx >> 2, kv = idx & 3;
      r[i] = *(const U4*)(Ab + (long long)(m0 + row) * K + kt * 32 + kv * 8);
    }
  };
  auto storeA = [&](int buf, const U4* r) {
#pragma unroll
    for (int i = 0; i < 2; ++i) {
      int idx = tid + i * 256, row = idx >> 2, kv = idx & 3;
      *(U4*)&As[buf][row * LDSS + kv * 8] = r[i];
    }
  };
  auto loadBt_regs = [&](int kt, U4* r) {   // B_TRANS: B[N,K]
#pragma unroll
    for (int i = 0; i < 2; ++i) {
      int idx = tid + i * 256, row = idx >> 2, kv = idx & 3;
      r[i] = *(const U4*)(Bb + (long long)(n0 + row) * K + kt * 32 + kv * 8);
    }
  };
  auto storeBt = [&](int buf, const U4* r) {
#pragma unroll
    for (int i = 0; i < 2; ++i) {
      int idx = tid + i * 256, row = idx >> 2, kv = idx & 3;
      *(U4*)&Bs[buf][row * LDSS + kv * 8] = r[i];
    }
  };
  auto loadBnt_regs = [&](int kt, U4* r) {  // !B_TRANS: B[K,N]
#pragma unroll
    for (int i = 0; i < 2; ++i) {
      int idx = tid + i * 256, krow = idx >> 4, nv = idx & 15;
      r[i] = *(const U4*)(Bb + (long long)(kt * 32 + krow) * N + n0 + nv * 8);
    }
  };
  auto storeBnt = [&](int buf, const U4* r) {  // transpose into [n][k]
#pragma unroll
    for (int i = 0; i < 2; ++i) {
      int idx = tid + i * 256, krow = idx >> 4, nv = idx & 15;
      const bf16* pv = (const bf16*)&r[i];
#pragma unroll
      for (int e = 0; e < 8; ++e) Bs[buf][(nv * 8 + e) * LDSS + krow] = pv[e];
    }
  };
#if HAVE_ASYNC_LDS
  auto asyncA = [&](int kt, int buf) {   // 16B/lane direct memory->LDS (ASYNCcnt)
#pragma unroll
    for (int i = 0; i < 2; ++i) {
      int idx = tid + i * 256, row = idx >> 2, kv = idx & 3;
      __builtin_amdgcn_global_load_async_to_lds_b128(
          TO_GLB(Ab + (long long)(m0 + row) * K + kt * 32 + kv * 8),
          TO_LDS(&As[buf][row * LDSS + kv * 8]), 0, 0);
    }
  };
  auto asyncBt = [&](int kt, int buf) {
#pragma unroll
    for (int i = 0; i < 2; ++i) {
      int idx = tid + i * 256, row = idx >> 2, kv = idx & 3;
      __builtin_amdgcn_global_load_async_to_lds_b128(
          TO_GLB(Bb + (long long)(n0 + row) * K + kt * 32 + kv * 8),
          TO_LDS(&Bs[buf][row * LDSS + kv * 8]), 0, 0);
    }
  };
#endif
  // ---- WMMA on resident tile: fragments per documented CDNA5 layouts ----
  auto compute = [&](int cur) {
    Frag a[4], b[2];
#pragma unroll
    for (int i = 0; i < 4; ++i) {
      int m = wm0 + i * 16 + l15;
      a[i].q[0] = *(const U4*)&As[cur][m * LDSS + khalf * 8];        // K = khalf*8..+7
      a[i].q[1] = *(const U4*)&As[cur][m * LDSS + 16 + khalf * 8];   // K = 16+khalf*8..+7
    }
#pragma unroll
    for (int j = 0; j < 2; ++j) {
      int n = wn0 + j * 16 + l15;
      b[j].q[0] = *(const U4*)&Bs[cur][n * LDSS + khalf * 8];        // K = khalf*8..+7
      b[j].q[1] = *(const U4*)&Bs[cur][n * LDSS + 16 + khalf * 8];   // K = 16+khalf*8..+7
    }
#pragma unroll
    for (int i = 0; i < 4; ++i)
#pragma unroll
      for (int j = 0; j < 2; ++j)
        acc[i][j] = __builtin_amdgcn_wmma_f32_16x16x32_bf16(
            false, a[i].v, false, b[j].v, (short)0, acc[i][j], false, false);
  };

  const int nk = K >> 5;

  if constexpr (ALIGNED) {
#if HAVE_ASYNC_LDS
    // ---------- async double-buffered pipeline ----------
    {
      asyncA(0, 0);
      if constexpr (B_TRANS) {
        asyncBt(0, 0);
      } else {
        U4 rb[2]; loadBnt_regs(0, rb); storeBnt(0, rb);
      }
      __builtin_amdgcn_s_wait_asynccnt(0);
      __syncthreads();
    }
    for (int t = 0; t < nk; ++t) {
      const int cur = t & 1, nxt = cur ^ 1;
      const bool hn = (t + 1 < nk);
      U4 rb[2];
      if (hn) {
        asyncA(t + 1, nxt);
        if constexpr (B_TRANS) asyncBt(t + 1, nxt);
        else                   loadBnt_regs(t + 1, rb);
      }
      compute(cur);                       // overlaps with in-flight async copies
      if (hn) {
        if constexpr (!B_TRANS) storeBnt(nxt, rb);
        __builtin_amdgcn_s_wait_asynccnt(0);
      }
      __syncthreads();
    }
#else
    // ---------- sync double-buffered pipeline (register staging) ----------
    {
      U4 ra[2], rb[2];
      loadA_regs(0, ra);
      if constexpr (B_TRANS) loadBt_regs(0, rb); else loadBnt_regs(0, rb);
      storeA(0, ra);
      if constexpr (B_TRANS) storeBt(0, rb); else storeBnt(0, rb);
      __syncthreads();
    }
    for (int t = 0; t < nk; ++t) {
      const int cur = t & 1, nxt = cur ^ 1;
      const bool hn = (t + 1 < nk);
      U4 ra[2], rb[2];
      if (hn) {
        loadA_regs(t + 1, ra);
        if constexpr (B_TRANS) loadBt_regs(t + 1, rb); else loadBnt_regs(t + 1, rb);
        __builtin_prefetch(Ab + (long long)(m0 + (tid >> 1)) * K + (t + 2) * 32, 0, 1);
      }
      compute(cur);
      if (hn) {
        storeA(nxt, ra);
        if constexpr (B_TRANS) storeBt(nxt, rb); else storeBnt(nxt, rb);
      }
      __syncthreads();
    }
#endif
  } else {
    // ---------- guarded single-buffer path (x_proj: N=40) ----------
    for (int k0 = 0; k0 < K; k0 += 32) {
#pragma unroll
      for (int i = 0; i < 2; ++i) {
        int idx = tid + i * 256, row = idx >> 2, kv = idx & 3;
        U4 val; val.u[0] = val.u[1] = val.u[2] = val.u[3] = 0u;
        int gm = m0 + row;
        if (gm < M) val = *(const U4*)(Ab + (long long)gm * K + (k0 + kv * 8));
        *(U4*)&As[0][row * LDSS + kv * 8] = val;
      }
#pragma unroll
      for (int i = 0; i < 2; ++i) {
        int idx = tid + i * 256, row = idx >> 2, kv = idx & 3;
        U4 val; val.u[0] = val.u[1] = val.u[2] = val.u[3] = 0u;
        int gn = n0 + row;
        if (gn < N) val = *(const U4*)(Bb + (long long)gn * K + (k0 + kv * 8));
        *(U4*)&Bs[0][row * LDSS + kv * 8] = val;
      }
      __syncthreads();
      compute(0);
      __syncthreads();
    }
  }

  // ---- epilogue ----
  float g = 1.0f;
  if (HAS_RESID && gate_ptr) g = *gate_ptr;
#pragma unroll
  for (int i = 0; i < 4; ++i)
#pragma unroll
    for (int j = 0; j < 2; ++j)
#pragma unroll
      for (int r = 0; r < 8; ++r) {
        int m = m0 + wm0 + i * 16 + khalf * 8 + r;   // C/D: M = r + 8*khalf
        int n = n0 + wn0 + j * 16 + l15;             // N = lane&15
        if (m < M && n < N) {
          float v = acc[i][j][r];
          if (bias) v += bias[n];
          if (ACT == 1) v = 0.5f * v * (1.0f + erff(v * 0.70710678118654752f));
          long long orow;
          if (REMAP) {  // m = bn*T + t  ->  (b*T+t)*N + node
            int bn = m / Tdim;  int t = m - bn * Tdim;
            int bb = bn / Nnodes; int node = bn - bb * Nnodes;
            orow = (long long)(bb * Tdim + t) * Nnodes + node;
          } else orow = m;
          long long oidx = bz * batchC + orow * (long long)N + n;
          if (HAS_RESID) v = resid[oidx] + g * v;
          if (STORE_BF16) ((bf16*)Cout)[oidx] = (bf16)v;
          else            ((float*)Cout)[oidx] = v;
        }
      }
}

// ---------------------------------------------------------------------------
// LayerNorm over D=128: one wave per row, 4 vals/lane, wave32 xor reductions.
// PERM: input row ((b*T+t)*N+node) written to output row (b*N+node)*T+t.
// ---------------------------------------------------------------------------
template<bool PERM>
__global__ __launch_bounds__(256) void layernorm_k(
    const float* __restrict__ x, const float* __restrict__ gam,
    const float* __restrict__ bet, bf16* __restrict__ out,
    int rows, int Tdim, int Nnodes)
{
  int row  = blockIdx.x * 8 + (threadIdx.x >> 5);
  int lane = threadIdx.x & 31;
  if (row >= rows) return;
  const float* xr = x + (long long)row * 128;
  float v[4]; float s = 0.f;
#pragma unroll
  for (int i = 0; i < 4; ++i) { v[i] = xr[lane + 32 * i]; s += v[i]; }
#pragma unroll
  for (int off = 16; off > 0; off >>= 1) s += __shfl_xor(s, off, 32);
  float mu = s * (1.0f / 128.0f);
  float var = 0.f;
#pragma unroll
  for (int i = 0; i < 4; ++i) { float d = v[i] - mu; var += d * d; }
#pragma unroll
  for (int off = 16; off > 0; off >>= 1) var += __shfl_xor(var, off, 32);
  float rs = rsqrtf(var * (1.0f / 128.0f) + 1e-5f);
  long long orow;
  if (PERM) {
    int b  = row / (Tdim * Nnodes);
    int r2 = row - b * (Tdim * Nnodes);
    int t  = r2 / Nnodes;
    int node = r2 - t * Nnodes;
    orow = (long long)(b * Nnodes + node) * Tdim + t;
  } else orow = row;
  bf16* o = out + orow * 128;
#pragma unroll
  for (int i = 0; i < 4; ++i) {
    int c = lane + 32 * i;
    o[c] = (bf16)((v[i] - mu) * rs * gam[c] + bet[c]);
  }
}

// Causal depthwise conv (K=4, left pad 3) on first 256 cols of xz, + SiLU.
__global__ __launch_bounds__(256) void conv_silu_k(
    const bf16* __restrict__ xz, const float* __restrict__ w,
    const float* __restrict__ cb, bf16* __restrict__ xc,
    int T, long long total)
{
  long long idx = (long long)blockIdx.x * 256 + threadIdx.x;
  if (idx >= total) return;
  int ch = (int)(idx & 255);
  long long row = idx >> 8;          // row = bn*T + t
  int t = (int)(row % T);
  float acc = cb[ch];
#pragma unroll
  for (int k = 0; k < 4; ++k) {
    int tt = t - 3 + k;
    if (tt >= 0) acc += (float)xz[(row - 3 + k) * 512 + ch] * w[ch * 4 + k];
  }
  acc = acc / (1.0f + __expf(-acc));  // SiLU
  xc[row * 256 + ch] = (bf16)acc;
}

// dt = softplus(x_dbl[:, :8] @ dt_proj_w.T + dt_proj_b)   (K=8 -> plain VALU)
__global__ __launch_bounds__(256) void dtproj_k(
    const float* __restrict__ xdbl, const float* __restrict__ w,
    const float* __restrict__ b, float* __restrict__ dt, long long total)
{
  long long idx = (long long)blockIdx.x * 256 + threadIdx.x;
  if (idx >= total) return;
  int d = (int)(idx & 255);
  long long row = idx >> 8;
  const float* xr = xdbl + row * 40;
  float acc = b[d];
#pragma unroll
  for (int j = 0; j < 8; ++j) acc += xr[j] * w[d * 8 + j];
  dt[idx] = (acc > 20.f) ? acc : log1pf(__expf(acc));
}

// Selective scan: one block per (b,node) sequence, one thread per d_inner
// channel; 16-wide state in registers, B_t/C_t broadcast via LDS.
// Fuses  y = (scan + D*u) * silu(z)  and stores bf16 for out_proj GEMM.
__global__ __launch_bounds__(256) void ssm_scan_k(
    const float* __restrict__ dt, const float* __restrict__ xdbl,
    const bf16* __restrict__ xc, const bf16* __restrict__ xz,
    const float* __restrict__ A_log, const float* __restrict__ Dp,
    bf16* __restrict__ y, int T)
{
  int bn = blockIdx.x;
  int d  = threadIdx.x;
  float Ar[16], h[16];
#pragma unroll
  for (int j = 0; j < 16; ++j) { Ar[j] = -__expf(A_log[d * 16 + j]); h[j] = 0.f; }
  float Dd = Dp[d];
  __shared__ float BC[32];
  long long base = (long long)bn * T;
  for (int t = 0; t < T; ++t) {
    long long row = base + t;
    if (d < 32) BC[d] = xdbl[row * 40 + 8 + d];  // B in [0..15], C in [16..31]
    __syncthreads();
    float dtv = dt[row * 256 + d];
    float u   = (float)xc[row * 256 + d];
    float zz  = (float)xz[row * 512 + 256 + d];
    float du  = dtv * u;
    float yv  = 0.f;
#pragma unroll
    for (int j = 0; j < 16; ++j) {
      h[j] = __expf(dtv * Ar[j]) * h[j] + du * BC[j];
      yv  += h[j] * BC[16 + j];
    }
    yv = (yv + Dd * u) * (zz / (1.0f + __expf(-zz)));
    y[row * 256 + d] = (bf16)yv;
    __syncthreads();
  }
}

// adj = softmax(relu(E1 @ E2^T), axis=-1), stored bf16. One block per row.
__global__ __launch_bounds__(256) void adj_k(
    const float* __restrict__ E1, const float* __restrict__ E2,
    bf16* __restrict__ adj)
{
  int n = blockIdx.x, tid = threadIdx.x;
  __shared__ float e1s[16];
  __shared__ float red[256];
  if (tid < 16) e1s[tid] = E1[n * 16 + tid];
  __syncthreads();
  float sc[2];
#pragma unroll
  for (int i = 0; i < 2; ++i) {
    int m = tid + i * 256;
    float s = 0.f;
#pragma unroll
    for (int j = 0; j < 16; ++j) s += e1s[j] * E2[m * 16 + j];
    sc[i] = fmaxf(s, 0.f);
  }
  red[tid] = fmaxf(sc[0], sc[1]);
  __syncthreads();
  for (int off = 128; off > 0; off >>= 1) {
    if (tid < off) red[tid] = fmaxf(red[tid], red[tid + off]);
    __syncthreads();
  }
  float mx = red[0];
  __syncthreads();
  float ea = __expf(sc[0] - mx), eb = __expf(sc[1] - mx);
  red[tid] = ea + eb;
  __syncthreads();
  for (int off = 128; off > 0; off >>= 1) {
    if (tid < off) red[tid] += red[tid + off];
    __syncthreads();
  }
  float inv = 1.0f / red[0];
  adj[n * 512 + tid]       = (bf16)(ea * inv);
  adj[n * 512 + tid + 256] = (bf16)(eb * inv);
}

__global__ void f2bf_k(const float* __restrict__ s, bf16* __restrict__ d, int n) {
  int i = blockIdx.x * 256 + threadIdx.x;
  if (i < n) d[i] = (bf16)s[i];
}

// ---------------------------------------------------------------------------
extern "C" void kernel_launch(void* const* d_in, const int* in_sizes, int n_in,
                              void* d_out, int out_size, void* d_ws, size_t ws_size,
                              hipStream_t stream)
{
  (void)in_sizes; (void)n_in; (void)out_size; (void)ws_size;
  constexpr int Bn = 4, T = 96, Nn = 512;
  const long long tokens = (long long)Bn * T * Nn;   // 196608
  const int Mtok = (int)tokens;

  const float* x        = (const float*)d_in[0];
  const float* in_proj  = (const float*)d_in[1];
  const float* conv_w   = (const float*)d_in[2];
  const float* conv_b   = (const float*)d_in[3];
  const float* x_proj   = (const float*)d_in[4];
  const float* dt_w     = (const float*)d_in[5];
  const float* dt_b     = (const float*)d_in[6];
  const float* A_log    = (const float*)d_in[7];
  const float* Dp       = (const float*)d_in[8];
  const float* out_proj = (const float*)d_in[9];
  const float* n1g = (const float*)d_in[10], *n1b = (const float*)d_in[11];
  const float* n2g = (const float*)d_in[12], *n2b = (const float*)d_in[13];
  const float* n3g = (const float*)d_in[14], *n3b = (const float*)d_in[15];
  const float* e1  = (const float*)d_in[16], *e2  = (const float*)d_in[17];
  const float* gc_w = (const float*)d_in[18], *gc_b = (const float*)d_in[19];
  const float* gate = (const float*)d_in[20];
  const float* f1w = (const float*)d_in[21], *f1b = (const float*)d_in[22];
  const float* f2w = (const float*)d_in[23], *f2b = (const float*)d_in[24];
  float* out = (float*)d_out;

  // ---- workspace bump allocation (256B aligned), with lifetime aliasing ----
  char* p = (char*)d_ws;
  auto alloc = [&](size_t bytes) -> char* {
    char* r = p; p += (bytes + 255) & ~(size_t)255; return r;
  };
  bf16* w_in  = (bf16*)alloc(65536 * 2);
  bf16* w_xp  = (bf16*)alloc(10240 * 2);
  bf16* w_out = (bf16*)alloc(32768 * 2);
  bf16* w_gc  = (bf16*)alloc(16384 * 2);
  bf16* w_f1  = (bf16*)alloc(65536 * 2);
  bf16* w_f2  = (bf16*)alloc(65536 * 2);
  bf16*  xnT  = (bf16*) alloc(tokens * 128 * 2);   // LN1 out, [B*N, T, D]
  bf16*  xzb  = (bf16*) alloc(tokens * 512 * 2);   // in_proj out; later FFN hidden
  bf16*  xcb  = (bf16*) alloc(tokens * 256 * 2);   // conv+silu out
  float* xdbl = (float*)alloc(tokens * 40  * 4);   // x_proj out
  float* dt   = (float*)alloc(tokens * 256 * 4);   // dt; later x2 (fp32)
  bf16*  yb   = (bf16*) alloc(tokens * 256 * 2);   // gated scan out; later xg
  float* x1   = (float*)alloc(tokens * 128 * 4);   // post-mamba residual
  bf16*  lnb  = (bf16*) alloc(tokens * 128 * 2);   // LN2 / LN3 out
  bf16*  adjb = (bf16*) alloc(512 * 512 * 2);
  float* x2 = dt;      // dt dead after scan
  bf16*  xg = yb;      // y dead after out_proj
  bf16*  h  = xzb;     // xz dead after scan (z consumed there)

  // weights fp32 -> bf16
  f2bf_k<<<256, 256, 0, stream>>>(in_proj,  w_in,  65536);
  f2bf_k<<<40,  256, 0, stream>>>(x_proj,   w_xp,  10240);
  f2bf_k<<<128, 256, 0, stream>>>(out_proj, w_out, 32768);
  f2bf_k<<<64,  256, 0, stream>>>(gc_w,     w_gc,  16384);
  f2bf_k<<<256, 256, 0, stream>>>(f1w,      w_f1,  65536);
  f2bf_k<<<256, 256, 0, stream>>>(f2w,      w_f2,  65536);

  // 1. LN1 (+ permute btnd -> (bn,t,d))
  layernorm_k<true><<<Mtok / 8, 256, 0, stream>>>(x, n1g, n1b, xnT, Mtok, T, Nn);
  // 2. xz = xn @ in_proj^T    [tokens,512] = [tokens,128] x [512,128]^T
  gemm_bf16_wmma<true, 0, true, false, false, true><<<dim3(4, Mtok / 128, 1), 256, 0, stream>>>(
      xnT, w_in, nullptr, nullptr, nullptr, xzb, Mtok, 512, 128, 0, 0, 0, 0, 0);
  // 3. xc = silu(causal_conv(xz[:, :256]))
  conv_silu_k<<<(unsigned)tokens, 256, 0, stream>>>(xzb, conv_w, conv_b, xcb, T, tokens * 256);
  // 4. x_dbl = xc @ x_proj^T   [tokens,40]  (N=40 -> guarded path)
  gemm_bf16_wmma<true, 0, false, false, false, false><<<dim3(1, Mtok / 128, 1), 256, 0, stream>>>(
      xcb, w_xp, nullptr, nullptr, nullptr, xdbl, Mtok, 40, 256, 0, 0, 0, 0, 0);
  // 5. dt = softplus(dt_proj)
  dtproj_k<<<(unsigned)tokens, 256, 0, stream>>>(xdbl, dt_w, dt_b, dt, tokens * 256);
  // 6. selective scan, fused with +D*u and *silu(z)
  ssm_scan_k<<<Bn * Nn, 256, 0, stream>>>(dt, xdbl, xcb, xzb, A_log, Dp, yb, T);
  // 7. x1 = x + y @ out_proj^T   (remap (bn,t) -> btnd, residual = original x)
  gemm_bf16_wmma<true, 0, false, true, true, true><<<dim3(1, Mtok / 128, 1), 256, 0, stream>>>(
      yb, w_out, nullptr, x, nullptr, x1, Mtok, 128, 256, 0, 0, 0, T, Nn);
  // 8. LN2
  layernorm_k<false><<<Mtok / 8, 256, 0, stream>>>(x1, n2g, n2b, lnb, Mtok, T, Nn);
  // 9. adjacency softmax(relu(E1 E2^T))
  adj_k<<<512, 256, 0, stream>>>(e1, e2, adjb);
  // 10. xg = adj @ xn2   batched over B*T (A fixed, B/C stride 512*128)
  gemm_bf16_wmma<false, 0, true, false, false, true><<<dim3(1, 4, Bn * T), 256, 0, stream>>>(
      adjb, lnb, nullptr, nullptr, nullptr, xg, 512, 128, 512,
      0, (long long)512 * 128, (long long)512 * 128, 0, 0);
  // 11. x2 = x1 + gate * (xg @ gc_w^T + gc_b)
  gemm_bf16_wmma<true, 0, false, false, true, true><<<dim3(1, Mtok / 128, 1), 256, 0, stream>>>(
      xg, w_gc, gc_b, x1, gate, x2, Mtok, 128, 128, 0, 0, 0, 0, 0);
  // 12. LN3
  layernorm_k<false><<<Mtok / 8, 256, 0, stream>>>(x2, n3g, n3b, lnb, Mtok, T, Nn);
  // 13. h = gelu(ln @ f1^T + b1)   [tokens,512]
  gemm_bf16_wmma<true, 1, true, false, false, true><<<dim3(4, Mtok / 128, 1), 256, 0, stream>>>(
      lnb, w_f1, f1b, nullptr, nullptr, h, Mtok, 512, 128, 0, 0, 0, 0, 0);
  // 14. out = x2 + h @ f2^T + b2
  gemm_bf16_wmma<true, 0, false, false, true, true><<<dim3(1, Mtok / 128, 1), 256, 0, stream>>>(
      h, w_f2, f2b, x2, nullptr, out, Mtok, 128, 512, 0, 0, 0, 0, 0);
}
